// psaa5NetHead_79989470921263
// MI455X (gfx1250) — matrix operand order repaired and checked by reading, not compiled, not run on hardware
//
#include <hip/hip_runtime.h>

// ============================================================================
// PSAA head for MI455X (gfx1250, wave32).
//  - All GEMMs use v_wmma_f32_16x16x32_bf16 (f32 accumulate).
//  - Mode-0 GEMMs: 64x64 block tile, 2x2 WMMA per wave, A/B staged to LDS via
//    GLOBAL_LOAD_ASYNC_TO_LDS_B128 (ASYNCcnt) with double buffering.
//  - PAM attention is flash-style (online softmax): no [N,N] scratch.
// ============================================================================

typedef __bf16 bf16;
typedef __attribute__((ext_vector_type(16))) __bf16 v16bf;
typedef __attribute__((ext_vector_type(8)))  __bf16 v8bf;
typedef __attribute__((ext_vector_type(8)))  float  v8f;
typedef int v4i_t __attribute__((vector_size(16)));  // builtin's pointee type

__device__ __forceinline__ v8f wmma_bf16(v16bf a, v16bf b, v8f c) {
  return __builtin_amdgcn_wmma_f32_16x16x32_bf16(false, a, false, b, (short)0, c,
                                                 false, false);
}

// ---- async global->LDS copy (16B per lane), guarded with sync fallback ----
__device__ __forceinline__ void cp_g2l_16B(const bf16* g, bf16* l) {
#if __has_builtin(__builtin_amdgcn_global_load_async_to_lds_b128)
  __builtin_amdgcn_global_load_async_to_lds_b128((v4i_t*)g, (v4i_t*)l, 0, 0);
#else
  *reinterpret_cast<v8bf*>(l) = *reinterpret_cast<const v8bf*>(g);
#endif
}
__device__ __forceinline__ void cp_wait_le4() {
#if __has_builtin(__builtin_amdgcn_global_load_async_to_lds_b128)
#if __has_builtin(__builtin_amdgcn_s_wait_asynccnt)
  __builtin_amdgcn_s_wait_asynccnt(4);
#else
  asm volatile("s_wait_asynccnt 4" ::: "memory");
#endif
#endif
}
__device__ __forceinline__ void cp_wait_le0() {
#if __has_builtin(__builtin_amdgcn_global_load_async_to_lds_b128)
#if __has_builtin(__builtin_amdgcn_s_wait_asynccnt)
  __builtin_amdgcn_s_wait_asynccnt(0);
#else
  asm volatile("s_wait_asynccnt 0" ::: "memory");
#endif
#endif
}

// ---------------------------------------------------------------------------
// f32 -> bf16 conversion
// ---------------------------------------------------------------------------
__global__ void f2b_kernel(const float* __restrict__ src, bf16* __restrict__ dst,
                           long long n) {
  long long i = (long long)blockIdx.x * 256 + threadIdx.x;
  if (i < n) dst[i] = (bf16)src[i];
}

// ---------------------------------------------------------------------------
// Mode-0 GEMM with async-LDS double-buffered staging.
// C[bz][M,N] = epi(A[bz][M,K] @ B[bz][K,N]); B row-major [K,N], N mult of 64.
// Block 128 thr = 4 waves (2x2); block tile 64M x 64N; wave tile 32x32
// (2x2 WMMA frags -> 2x A reuse, 2x B reuse per k-step).
// ---------------------------------------------------------------------------
__global__ void __launch_bounds__(128) gemm0_kernel(
    const bf16* __restrict__ A, int lda, long long strideA,
    const bf16* __restrict__ B, long long strideB, int ldb,
    int M, int N, int K,
    const float* __restrict__ scale, const float* __restrict__ bias,
    const float* __restrict__ alpha_p,
    const float* __restrict__ resid, long long strideR, int do_relu,
    float* __restrict__ outF, long long strideF,
    bf16* __restrict__ outB, long long strideOB) {
  const int tid  = threadIdx.x;
  const int lane = tid & 31, wvid = tid >> 5;
  const int wm = wvid >> 1, wn = wvid & 1;
  const int nbase0 = blockIdx.x * 64;
  const int mbase0 = blockIdx.y * 64;
  const int bz = blockIdx.z;
  const bf16* Ap = A + (long long)bz * strideA;
  const bf16* Bp = B + (long long)bz * strideB + nbase0;
  const int ml = lane & 15, hi = lane >> 4;

  __shared__ bf16 s_a[2][64][32];  // [buf][m][k]
  __shared__ bf16 s_b[2][32][64];  // [buf][k][n]

  // stage one 64x32 A tile + 32x64 B tile (4 async issues per thread)
  auto stage = [&](int kt, int buf) {
    const int k0 = kt * 32;
#pragma unroll
    for (int r = 0; r < 2; ++r) {
      int slot = tid + r * 128;            // A: 256 slots of 16B
      int row = slot >> 2, seg = slot & 3;
      cp_g2l_16B(Ap + (long long)(mbase0 + row) * lda + k0 + seg * 8,
                 &s_a[buf][row][seg * 8]);
    }
#pragma unroll
    for (int r = 0; r < 2; ++r) {
      int slot = tid + r * 128;            // B: 256 slots of 16B
      int row = slot >> 3, seg = slot & 7;
      cp_g2l_16B(Bp + (long long)(k0 + row) * ldb + seg * 8,
                 &s_b[buf][row][seg * 8]);
    }
  };

  v8f acc[2][2];
#pragma unroll
  for (int sm = 0; sm < 2; ++sm)
#pragma unroll
    for (int sn = 0; sn < 2; ++sn)
#pragma unroll
      for (int e = 0; e < 8; ++e) acc[sm][sn][e] = 0.f;

  const int KT = K >> 5;
  stage(0, 0);
  for (int kt = 0; kt < KT; ++kt) {
    const int buf = kt & 1;
    if (kt + 1 < KT) {
      stage(kt + 1, buf ^ 1);
      cp_wait_le4();   // allow next-stage 4 issues in flight; this stage done
    } else {
      cp_wait_le0();
    }
    __syncthreads();

    v16bf afr[2];
#pragma unroll
    for (int sm = 0; sm < 2; ++sm) {
      int row = wm * 32 + sm * 16 + ml;
      int kb = hi ? 16 : 0;
      v8bf lo = *reinterpret_cast<const v8bf*>(&s_a[buf][row][kb]);
      v8bf hv = *reinterpret_cast<const v8bf*>(&s_a[buf][row][kb + 8]);
#pragma unroll
      for (int e = 0; e < 8; ++e) { afr[sm][e] = lo[e]; afr[sm][8 + e] = hv[e]; }
    }
    v16bf bfr[2];
#pragma unroll
    for (int sn = 0; sn < 2; ++sn) {
      int c = wn * 32 + sn * 16 + ml;
      int kk = hi ? 16 : 0;
#pragma unroll
      for (int e = 0; e < 16; ++e) bfr[sn][e] = s_b[buf][kk + e][c];
    }
#pragma unroll
    for (int sm = 0; sm < 2; ++sm)
#pragma unroll
      for (int sn = 0; sn < 2; ++sn)
        acc[sm][sn] = wmma_bf16(afr[sm], bfr[sn], acc[sm][sn]);
    __syncthreads();  // reads done before next overwrite
  }

  const float al = alpha_p ? alpha_p[0] : 1.0f;
#pragma unroll
  for (int sm = 0; sm < 2; ++sm) {
#pragma unroll
    for (int sn = 0; sn < 2; ++sn) {
#pragma unroll
      for (int e = 0; e < 8; ++e) {
        int m = mbase0 + wm * 32 + sm * 16 + e + (hi ? 8 : 0);
        if (m >= M) continue;
        int n = nbase0 + wn * 32 + sn * 16 + ml;
        float y = acc[sm][sn][e];
        if (scale) y *= scale[m];
        if (bias) y += bias[m];
        y *= al;
        if (do_relu) y = fmaxf(y, 0.f);
        long long idx = (long long)m * N + n;
        if (resid) y += resid[(long long)bz * strideR + idx];
        if (outF) outF[(long long)bz * strideF + idx] = y;
        if (outB) outB[(long long)bz * strideOB + idx] = (bf16)y;
      }
    }
  }
}

// ---------------------------------------------------------------------------
// Direct-fragment GEMM for special B layouts.
// BMODE 1: B transposed: B[k][n] = Bsrc[n*ldb + k]
// BMODE 2: dilated-3x3 gather: K = 512*9, B[k][n] from shifted h (rate)
// Block 128 thr = 4 waves; block tile 16M x 64N; each wave 16x16.
// ---------------------------------------------------------------------------
template <int BMODE>
__global__ void __launch_bounds__(128) gemm_kernel(
    const bf16* __restrict__ A, int lda, long long strideA,
    const bf16* __restrict__ B, long long strideB, int ldb,
    int M, int N, int K, int rate,
    const float* __restrict__ scale, const float* __restrict__ bias,
    const float* __restrict__ alpha_p,
    const float* __restrict__ resid, long long strideR, int do_relu,
    float* __restrict__ outF, long long strideF,
    bf16* __restrict__ outB, long long strideOB) {
  const int lane  = threadIdx.x & 31;
  const int wv    = threadIdx.x >> 5;
  const int nbase = blockIdx.x * 64 + wv * 16;
  const int mbase = blockIdx.y * 16;
  const int bz    = blockIdx.z;
  const bf16* Ap = A + (long long)bz * strideA;
  const bf16* Bp = B + (long long)bz * strideB;
  const int mloc = lane & 15;
  const int hi   = (lane >> 4) & 1;
  const int mrow = mbase + mloc;
  const int ncol = nbase + mloc;

  v8f acc;
#pragma unroll
  for (int e = 0; e < 8; ++e) acc[e] = 0.f;

  for (int k0 = 0; k0 < K; k0 += 32) {
    v16bf a;
    {
      int kb = k0 + (hi ? 16 : 0);
      if (mrow < M) {
        const bf16* ap = Ap + (long long)mrow * lda + kb;
        v8bf lo = *reinterpret_cast<const v8bf*>(ap);
        v8bf hv = *reinterpret_cast<const v8bf*>(ap + 8);
#pragma unroll
        for (int e = 0; e < 8; ++e) { a[e] = lo[e]; a[8 + e] = hv[e]; }
      } else {
#pragma unroll
        for (int e = 0; e < 16; ++e) a[e] = (bf16)0.f;
      }
    }
    v16bf bf;
    const int kk = k0 + (hi ? 16 : 0);
    if (BMODE == 1) {
      const bf16* bp = Bp + (long long)ncol * ldb + kk;
      bf = *reinterpret_cast<const v16bf*>(bp);
    } else {
#pragma unroll
      for (int e = 0; e < 16; ++e) {
        int kg = kk + e;
        int i = kg / 9, tap = kg - i * 9;
        int dy = tap / 3 - 1, dx = tap - (tap / 3) * 3 - 1;
        int ys = (ncol >> 6) + dy * rate;
        int xs = (ncol & 63) + dx * rate;
        bf[e] = ((unsigned)ys < 64u && (unsigned)xs < 64u)
                    ? Bp[(long long)i * 4096 + (ys << 6) + xs]
                    : (bf16)0.f;
      }
    }
    acc = wmma_bf16(a, bf, acc);
  }

  const float al = alpha_p ? alpha_p[0] : 1.0f;
#pragma unroll
  for (int e = 0; e < 8; ++e) {
    int m = mbase + e + (hi ? 8 : 0);
    if (m >= M) continue;
    float y = acc[e];
    if (scale) y *= scale[m];
    if (bias) y += bias[m];
    y *= al;
    if (do_relu) y = fmaxf(y, 0.f);
    long long idx = (long long)m * N + ncol;
    if (resid) y += resid[(long long)bz * strideR + idx];
    if (outF) outF[(long long)bz * strideF + idx] = y;
    if (outB) outB[(long long)bz * strideOB + idx] = (bf16)y;
  }
}

// ---------------------------------------------------------------------------
// Global average pool over 4096 pixels, one block per (b,c) of x [2,2048,4096]
// ---------------------------------------------------------------------------
__global__ void gap_kernel(const float* __restrict__ x, float* __restrict__ gap) {
  int bc = blockIdx.x, t = threadIdx.x;
  const float* p = x + ((long long)bc << 12);
  float s = 0.f;
  for (int i = t; i < 4096; i += 256) s += p[i];
  __shared__ float red[256];
  red[t] = s;
  __syncthreads();
  for (int off = 128; off; off >>= 1) {
    if (t < off) red[t] += red[t + off];
    __syncthreads();
  }
  if (t == 0) gap[bc] = red[0] * (1.0f / 4096.0f);
}

__global__ void b4_kernel(const float* __restrict__ w, const float* __restrict__ s4,
                          const float* __restrict__ bb,
                          const float* __restrict__ gap,
                          float* __restrict__ feat4) {
  int b = blockIdx.x, c = threadIdx.x;
  const float* wr = w + (long long)c * 2048;
  const float* g  = gap + (long long)b * 2048;
  float acc = 0.f;
  for (int i = 0; i < 2048; ++i) acc += wr[i] * g[i];
  feat4[b * 256 + c] = fmaxf(s4[c] * acc + bb[c], 0.f);
}

__global__ void bcast_kernel(const float* __restrict__ feat4,
                             bf16* __restrict__ feats) {
  int bc = blockIdx.x;
  int b = bc >> 8, c = bc & 255;
  bf16 v = (bf16)feat4[bc];
  bf16* dst = feats + ((((long long)b * 5 + 4) * 256 + c) << 12);
  for (int i = threadIdx.x; i < 4096; i += 256) dst[i] = v;
}

// ---------------------------------------------------------------------------
// PSAA per-pixel 5-way attention. One block (256 thr = channels) per pixel.
// ---------------------------------------------------------------------------
__global__ void psaa_kernel(const float* __restrict__ query,
                            const bf16* __restrict__ feats,
                            const float* __restrict__ gamma_p,
                            float* __restrict__ out1f, bf16* __restrict__ out1b) {
  int n = blockIdx.x & 4095;
  int b = blockIdx.x >> 12;
  int c = threadIdx.x;
  __shared__ float red[256];
  __shared__ float esh[5];
  __shared__ float ash[5];
  long long qidx = (((long long)b * 256 + c) << 12) + n;
  float q = query[qidx];
  float ys[5];
#pragma unroll
  for (int s = 0; s < 5; ++s)
    ys[s] = (float)feats[((((long long)b * 5 + s) * 256 + c) << 12) + n];
  for (int s = 0; s < 5; ++s) {
    red[c] = q * ys[s];
    __syncthreads();
    for (int off = 128; off; off >>= 1) {
      if (c < off) red[c] += red[c + off];
      __syncthreads();
    }
    if (c == 0) esh[s] = red[0];
    __syncthreads();
  }
  if (c == 0) {
    float mx = esh[0];
    for (int s = 1; s < 5; ++s) mx = fmaxf(mx, esh[s]);
    float en[5], mx2 = -1e30f;
    for (int s = 0; s < 5; ++s) { en[s] = mx - esh[s]; mx2 = fmaxf(mx2, en[s]); }
    float sum = 0.f;
    for (int s = 0; s < 5; ++s) { ash[s] = __expf(en[s] - mx2); sum += ash[s]; }
    float inv = 1.0f / sum;
    for (int s = 0; s < 5; ++s) ash[s] *= inv;
  }
  __syncthreads();
  float o = 0.f;
#pragma unroll
  for (int s = 0; s < 5; ++s) o += ash[s] * ys[s];
  float val = gamma_p[0] * o + q;
  out1f[qidx] = val;
  out1b[qidx] = (bf16)val;
}

// ---------------------------------------------------------------------------
// Flash-attention PAM. q,k: [b][64,4096]; v,out: [b][256,4096] (bf16).
// One wave = 16-row i-tile; streams 32-wide j tiles with online softmax.
// V fragments batched into registers (one wait) then 16 back-to-back WMMAs.
// ---------------------------------------------------------------------------
__global__ void __launch_bounds__(128) flash_pam_kernel(
    const bf16* __restrict__ qb, const bf16* __restrict__ kb,
    const bf16* __restrict__ vb, const float* __restrict__ gamma_p,
    const float* __restrict__ out1f, float* __restrict__ out2f,
    bf16* __restrict__ out2b) {
  const int lane = threadIdx.x & 31;
  const int wv   = threadIdx.x >> 5;
  const int b    = blockIdx.y;
  const int ibase = blockIdx.x * 64 + wv * 16;
  const long long qoff = (long long)b * 64 * 4096;
  const long long coff = (long long)b * 256 * 4096;
  const bf16* q = qb + qoff;
  const bf16* k = kb + qoff;
  const bf16* v = vb + coff;

  __shared__ float s_s[4][16][32];
  __shared__ bf16  s_p[4][16][32];
  __shared__ float s_sc[4][16];

  const int m  = lane & 15;
  const int hi = lane >> 4;

  v16bf qa0, qa1;
  {
    int kb0 = hi ? 16 : 0;
#pragma unroll
    for (int e = 0; e < 8; ++e) {
      qa0[e]     = q[(long long)(kb0 + e) * 4096 + ibase + m];
      qa0[8 + e] = q[(long long)(kb0 + 8 + e) * 4096 + ibase + m];
      qa1[e]     = q[(long long)(32 + kb0 + e) * 4096 + ibase + m];
      qa1[8 + e] = q[(long long)(32 + kb0 + 8 + e) * 4096 + ibase + m];
    }
  }

  v8f O[16];
#pragma unroll
  for (int t = 0; t < 16; ++t)
#pragma unroll
    for (int e = 0; e < 8; ++e) O[t][e] = 0.f;

  float run_m = -1e30f, run_l = 0.f;
  const int kk = hi ? 16 : 0;

  for (int j0 = 0; j0 < 4096; j0 += 32) {
    if (j0 + 32 < 4096)
      __builtin_prefetch(k + (long long)kk * 4096 + j0 + 32, 0, 1);
    v8f S0, S1;
#pragma unroll
    for (int e = 0; e < 8; ++e) { S0[e] = 0.f; S1[e] = 0.f; }
    {
      v16bf kf;
      int col0 = j0 + m, col1 = j0 + 16 + m;
#pragma unroll
      for (int e = 0; e < 16; ++e) kf[e] = k[(long long)(kk + e) * 4096 + col0];
      S0 = wmma_bf16(qa0, kf, S0);
#pragma unroll
      for (int e = 0; e < 16; ++e) kf[e] = k[(long long)(32 + kk + e) * 4096 + col0];
      S0 = wmma_bf16(qa1, kf, S0);
#pragma unroll
      for (int e = 0; e < 16; ++e) kf[e] = k[(long long)(kk + e) * 4096 + col1];
      S1 = wmma_bf16(qa0, kf, S1);
#pragma unroll
      for (int e = 0; e < 16; ++e) kf[e] = k[(long long)(32 + kk + e) * 4096 + col1];
      S1 = wmma_bf16(qa1, kf, S1);
    }
#pragma unroll
    for (int e = 0; e < 8; ++e) {
      int r = e + (hi ? 8 : 0);
      s_s[wv][r][m]      = S0[e];
      s_s[wv][r][16 + m] = S1[e];
    }
    __syncthreads();
    float vals[16], mx = -1e30f;
#pragma unroll
    for (int t = 0; t < 16; ++t) {
      vals[t] = s_s[wv][m][hi * 16 + t];
      mx = fmaxf(mx, vals[t]);
    }
    mx = fmaxf(mx, __shfl_xor(mx, 16, 32));
    float mnew  = fmaxf(run_m, mx);
    float alpha = __expf(run_m - mnew);
    float ls = 0.f;
#pragma unroll
    for (int t = 0; t < 16; ++t) {
      float p = __expf(vals[t] - mnew);
      ls += p;
      s_p[wv][m][hi * 16 + t] = (bf16)p;
    }
    ls += __shfl_xor(ls, 16, 32);
    run_l = run_l * alpha + ls;
    run_m = mnew;
    if (!hi) s_sc[wv][m] = alpha;
    __syncthreads();
    // batched V fragment loads: one clause of global_load_b128s, one wait
    v16bf vf[16];
#pragma unroll
    for (int ct = 0; ct < 16; ++ct)
      vf[ct] = *reinterpret_cast<const v16bf*>(
          v + (long long)(ct * 16 + m) * 4096 + j0 + kk);
    float sce[8];
#pragma unroll
    for (int e = 0; e < 8; ++e) sce[e] = s_sc[wv][e + (hi ? 8 : 0)];
#pragma unroll
    for (int t = 0; t < 16; ++t)
#pragma unroll
      for (int e = 0; e < 8; ++e) O[t][e] *= sce[e];
    v16bf pa;
    {
      int kb2 = hi ? 16 : 0;
#pragma unroll
      for (int e = 0; e < 8; ++e) {
        pa[e]     = s_p[wv][m][kb2 + e];
        pa[8 + e] = s_p[wv][m][kb2 + 8 + e];
      }
    }
#pragma unroll
    for (int ct = 0; ct < 16; ++ct) O[ct] = wmma_bf16(pa, vf[ct], O[ct]);
    __syncthreads();
  }

  if (!hi) s_sc[wv][m] = 1.0f / run_l;
  __syncthreads();
  float inv[8];
#pragma unroll
  for (int e = 0; e < 8; ++e) inv[e] = s_sc[wv][e + (hi ? 8 : 0)];
  const float g = gamma_p[0];
#pragma unroll
  for (int ct = 0; ct < 16; ++ct) {
#pragma unroll
    for (int e = 0; e < 8; ++e) {
      int i = ibase + e + (hi ? 8 : 0);
      int c = ct * 16 + m;
      long long idx = coff + (long long)c * 4096 + i;
      float val = g * (O[ct][e] * inv[e]) + out1f[idx];
      out2f[idx] = val;
      out2b[idx] = (bf16)val;
    }
  }
}

// ---------------------------------------------------------------------------
// CAM row softmax with max-minus trick: att = softmax(rowmax - e)
// ---------------------------------------------------------------------------
__global__ void cam_softmax_kernel(const float* __restrict__ E,
                                   bf16* __restrict__ att) {
  int row = blockIdx.x, t = threadIdx.x;
  const float* e = E + (long long)row * 256;
  __shared__ float red[256];
  float val = e[t];
  red[t] = val;
  __syncthreads();
  for (int off = 128; off; off >>= 1) {
    if (t < off) red[t] = fmaxf(red[t], red[t + off]);
    __syncthreads();
  }
  float mx = red[0];
  __syncthreads();
  float en = mx - val;
  red[t] = en;
  __syncthreads();
  for (int off = 128; off; off >>= 1) {
    if (t < off) red[t] = fmaxf(red[t], red[t + off]);
    __syncthreads();
  }
  float mx2 = red[0];
  __syncthreads();
  float p = __expf(en - mx2);
  red[t] = p;
  __syncthreads();
  for (int off = 128; off; off >>= 1) {
    if (t < off) red[t] += red[t + off];
    __syncthreads();
  }
  att[(long long)row * 256 + t] = (bf16)(p / red[0]);
}

// ===========================================================================
// Host-side orchestration
// ===========================================================================
// d_in flatten order (jax pytree: dict keys sorted at each level):
enum {
  I_B0_B1 = 0, I_B0_B2, I_B0_S1, I_B0_S2, I_B0_W1, I_B0_W2,
  I_B1_B1, I_B1_B2, I_B1_S1, I_B1_S2, I_B1_W1, I_B1_W2,
  I_B2_B1, I_B2_B2, I_B2_S1, I_B2_S2, I_B2_W1, I_B2_W2,
  I_B3_B1, I_B3_B2, I_B3_S1, I_B3_S2, I_B3_W1, I_B3_W2,
  I_B4_B, I_B4_S, I_B4_W,
  I_CAM_GAMMA,
  I_C8_B, I_C8_W,
  I_GAMMA,
  I_PAM_BK, I_PAM_BQ, I_PAM_BV, I_PAM_GAMMA, I_PAM_WK, I_PAM_WQ, I_PAM_WV,
  I_PROJ_B, I_PROJ_S, I_PROJ_W,
  I_X
};

extern "C" void kernel_launch(void* const* d_in, const int* in_sizes, int n_in,
                              void* d_out, int out_size, void* d_ws,
                              size_t ws_size, hipStream_t stream) {
  (void)in_sizes; (void)n_in; (void)out_size; (void)ws_size;
#define FP(i) ((const float*)d_in[i])

  // ---- workspace carve (all offsets 256B aligned); total ~135 MB ----
  char* wp = (char*)d_ws;
  auto carve = [&](size_t bytes) -> void* {
    void* p = (void*)wp;
    wp += (bytes + 255) & ~(size_t)255;
    return p;
  };
  bf16* xb = (bf16*)carve(16777216ull * 2);  // x bf16 [2,2048,4096]
  bf16* w1b[4];
  for (int i = 0; i < 4; ++i) w1b[i] = (bf16*)carve(1048576ull * 2);
  bf16* w20b   = (bf16*)carve(131072ull * 2);
  bf16* w2db[3];
  for (int i = 0; i < 3; ++i) w2db[i] = (bf16*)carve(1179648ull * 2);
  bf16* projwb = (bf16*)carve(327680ull * 2);
  bf16* wqb    = (bf16*)carve(16384ull * 2);
  bf16* wkb    = (bf16*)carve(16384ull * 2);
  bf16* wvb    = (bf16*)carve(65536ull * 2);
  bf16* w8b    = (bf16*)carve(4864ull * 2);
  bf16* hb     = (bf16*)carve(4194304ull * 2);   // [2,512,4096]
  bf16* featsb = (bf16*)carve(10485760ull * 2);  // [2,5,256,4096]
  float* queryf = (float*)carve(2097152ull * 4);
  float* out1f  = (float*)carve(2097152ull * 4);
  bf16*  out1b  = (bf16*)carve(2097152ull * 2);
  bf16*  qbuf   = (bf16*)carve(524288ull * 2);   // [2,64,4096]
  bf16*  kbuf   = (bf16*)carve(524288ull * 2);
  bf16*  vbuf   = (bf16*)carve(2097152ull * 2);
  float* out2f  = (float*)carve(2097152ull * 4);
  bf16*  out2b  = (bf16*)carve(2097152ull * 2);
  float* camE   = (float*)carve(131072ull * 4);  // [2,256,256]
  bf16*  camA   = (bf16*)carve(131072ull * 2);
  float* out3f  = (float*)carve(2097152ull * 4);
  bf16*  out3b  = (bf16*)carve(2097152ull * 2);
  float* gapf   = (float*)carve(4096ull * 4);
  float* feat4f = (float*)carve(512ull * 4);

  auto cvt = [&](int idx, bf16* dst, long long n) {
    f2b_kernel<<<dim3((unsigned)((n + 255) / 256)), dim3(256), 0, stream>>>(
        FP(idx), dst, n);
  };
  cvt(I_X, xb, 16777216);
  cvt(I_B0_W1, w1b[0], 1048576);
  cvt(I_B1_W1, w1b[1], 1048576);
  cvt(I_B2_W1, w1b[2], 1048576);
  cvt(I_B3_W1, w1b[3], 1048576);
  cvt(I_B0_W2, w20b, 131072);
  cvt(I_B1_W2, w2db[0], 1179648);
  cvt(I_B2_W2, w2db[1], 1179648);
  cvt(I_B3_W2, w2db[2], 1179648);
  cvt(I_PROJ_W, projwb, 327680);
  cvt(I_PAM_WQ, wqb, 16384);
  cvt(I_PAM_WK, wkb, 16384);
  cvt(I_PAM_WV, wvb, 65536);
  cvt(I_C8_W, w8b, 4864);

  const dim3 blk(128);

  // ---- global-pool branch ----
  gap_kernel<<<dim3(4096), dim3(256), 0, stream>>>(FP(I_X), gapf);
  b4_kernel<<<dim3(2), dim3(256), 0, stream>>>(FP(I_B4_W), FP(I_B4_S),
                                               FP(I_B4_B), gapf, feat4f);
  bcast_kernel<<<dim3(512), dim3(256), 0, stream>>>(feat4f, featsb);

  // ---- ASPP branches ----
  const int s1i[4] = {I_B0_S1, I_B1_S1, I_B2_S1, I_B3_S1};
  const int b1i[4] = {I_B0_B1, I_B1_B1, I_B2_B1, I_B3_B1};
  const int s2i[4] = {I_B0_S2, I_B1_S2, I_B2_S2, I_B3_S2};
  const int b2i[4] = {I_B0_B2, I_B1_B2, I_B2_B2, I_B3_B2};
  const int rates[3] = {12, 24, 36};
  for (int i = 0; i < 4; ++i) {
    // h = relu(bn(w1 @ x))  [512,4096] bf16
    gemm0_kernel<<<dim3(64, 8, 2), blk, 0, stream>>>(
        w1b[i], 2048, 0, xb, 8388608LL, 4096, 512, 4096, 2048,
        FP(s1i[i]), FP(b1i[i]), nullptr, nullptr, 0, 1,
        nullptr, 0, hb, 2097152LL);
    if (i == 0) {
      gemm0_kernel<<<dim3(64, 4, 2), blk, 0, stream>>>(
          w20b, 512, 0, hb, 2097152LL, 4096, 256, 4096, 512,
          FP(s2i[i]), FP(b2i[i]), nullptr, nullptr, 0, 1,
          nullptr, 0, featsb, 5242880LL);
    } else {
      gemm_kernel<2><<<dim3(64, 16, 2), blk, 0, stream>>>(
          w2db[i - 1], 4608, 0, hb, 2097152LL, 4096, 256, 4096, 4608,
          rates[i - 1], FP(s2i[i]), FP(b2i[i]), nullptr, nullptr, 0, 1,
          nullptr, 0, featsb + (long long)i * 1048576, 5242880LL);
    }
  }

  // ---- proj: query = relu(bn(Wp @ concat(feats))) ----
  gemm0_kernel<<<dim3(64, 4, 2), blk, 0, stream>>>(
      projwb, 1280, 0, featsb, 5242880LL, 4096, 256, 4096, 1280,
      FP(I_PROJ_S), FP(I_PROJ_B), nullptr, nullptr, 0, 1,
      queryf, 1048576LL, nullptr, 0);

  // ---- PSAA 5-way attention ----
  psaa_kernel<<<dim3(8192), dim3(256), 0, stream>>>(queryf, featsb, FP(I_GAMMA),
                                                    out1f, out1b);

  // ---- PAM q/k/v ----
  gemm0_kernel<<<dim3(64, 1, 2), blk, 0, stream>>>(
      wqb, 256, 0, out1b, 1048576LL, 4096, 64, 4096, 256,
      nullptr, FP(I_PAM_BQ), nullptr, nullptr, 0, 0,
      nullptr, 0, qbuf, 262144LL);
  gemm0_kernel<<<dim3(64, 1, 2), blk, 0, stream>>>(
      wkb, 256, 0, out1b, 1048576LL, 4096, 64, 4096, 256,
      nullptr, FP(I_PAM_BK), nullptr, nullptr, 0, 0,
      nullptr, 0, kbuf, 262144LL);
  gemm0_kernel<<<dim3(64, 4, 2), blk, 0, stream>>>(
      wvb, 256, 0, out1b, 1048576LL, 4096, 256, 4096, 256,
      nullptr, FP(I_PAM_BV), nullptr, nullptr, 0, 0,
      nullptr, 0, vbuf, 1048576LL);

  // ---- PAM flash attention ----
  flash_pam_kernel<<<dim3(64, 2), blk, 0, stream>>>(
      qbuf, kbuf, vbuf, FP(I_PAM_GAMMA), out1f, out2f, out2b);

  // ---- CAM ----
  gemm_kernel<1><<<dim3(4, 16, 2), blk, 0, stream>>>(
      out2b, 4096, 1048576LL, out2b, 1048576LL, 4096, 256, 256, 4096, 0,
      nullptr, nullptr, nullptr, nullptr, 0, 0,
      camE, 65536LL, nullptr, 0);
  cam_softmax_kernel<<<dim3(512), dim3(256), 0, stream>>>(camE, camA);
  gemm0_kernel<<<dim3(64, 4, 2), blk, 0, stream>>>(
      camA, 256, 65536LL, out2b, 1048576LL, 4096, 256, 4096, 256,
      nullptr, nullptr, FP(I_CAM_GAMMA), out2f, 1048576LL, 0,
      out3f, 1048576LL, out3b, 1048576LL);

  // ---- classifier: logits[b,19,4096] ----
  gemm0_kernel<<<dim3(64, 1, 2), blk, 0, stream>>>(
      w8b, 256, 0, out3b, 1048576LL, 4096, 19, 4096, 256,
      nullptr, FP(I_C8_B), nullptr, nullptr, 0, 0,
      (float*)d_out, 77824LL, nullptr, 0);
#undef FP
}
